// SCN_49478023250099
// MI455X (gfx1250) — matrix operand order repaired and checked by reading, not compiled, not run on hardware
//
#include <hip/hip_runtime.h>
#include <hip/hip_bf16.h>

typedef __attribute__((ext_vector_type(16))) _Float16 v16h;
typedef __attribute__((ext_vector_type(8)))  _Float16 v8h;
typedef __attribute__((ext_vector_type(8)))  float    v8f;

#define FDIM 256   // inner dim (features)
#define ODIM 256   // output cols
#define BM   64    // rows per block in GEMM
#define KS   32    // K-slab per WMMA stage

// ---------------------------------------------------------------------------
// Zero the output buffer (float4 stores).
// ---------------------------------------------------------------------------
__global__ void zero_f32(float* __restrict__ p, long n4) {
    long i = (long)blockIdx.x * blockDim.x + threadIdx.x;
    if (i < n4) {
        float4 z = {0.f, 0.f, 0.f, 0.f};
        ((float4*)p)[i] = z;
    }
}

// ---------------------------------------------------------------------------
// Y = x @ theta  (fp32 via split hi/lo f16 WMMA, fp32 accumulation)
// Block: 256 threads = 8 waves. Each block: 64 rows x 256 cols.
// LDS: Xh/Xl [64][256] halves, Th/Tl transposed [256][32] halves (96 KB).
// ---------------------------------------------------------------------------
__global__ void __launch_bounds__(256)
gemm_x_theta(const float* __restrict__ x, const float* __restrict__ theta,
             float* __restrict__ Y, int nrows) {
    extern __shared__ char smem[];
    _Float16* Xh = (_Float16*)smem;            // [BM][FDIM]
    _Float16* Xl = Xh + BM * FDIM;             // [BM][FDIM]
    _Float16* Th = Xl + BM * FDIM;             // transposed: [ODIM][KS]
    _Float16* Tl = Th + ODIM * KS;             // transposed: [ODIM][KS]

    const int tid  = threadIdx.x;
    const int lane = tid & 31;
    const int wave = tid >> 5;                 // 0..7
    const int row0 = blockIdx.x * BM;

    // ---- Stage X rows -> hi/lo f16 in LDS (float4 global loads) ----
    for (int i = tid; i < BM * FDIM / 4; i += 256) {
        int r  = (i * 4) / FDIM;
        int cc = (i * 4) % FDIM;
        float4 v = {0.f, 0.f, 0.f, 0.f};
        int gr = row0 + r;
        if (gr < nrows) v = *(const float4*)(x + (size_t)gr * FDIM + cc);
        float a[4] = {v.x, v.y, v.z, v.w};
        #pragma unroll
        for (int q = 0; q < 4; ++q) {
            _Float16 h = (_Float16)a[q];
            Xh[r * FDIM + cc + q] = h;
            Xl[r * FDIM + cc + q] = (_Float16)(a[q] - (float)h);
        }
    }

    // Wave tile assignment: 4 M-tiles x 16 N-tiles = 64 tiles, 8 per wave.
    const int mt     = wave >> 1;              // M-tile 0..3
    const int ntbase = (wave & 1) * 8;         // N-tiles [ntbase, ntbase+8)

    v8f acc[8];
    #pragma unroll
    for (int t = 0; t < 8; ++t)
        #pragma unroll
        for (int r = 0; r < 8; ++r) acc[t][r] = 0.0f;

    // A-fragment lane addressing (contiguous k order used here; B loads match):
    // lane 0-15: row=lane,    halves 0-15 -> K=0..15
    // lane 16-31: row=lane-16, halves 0-15 -> K=16..31
    const int am = mt * 16 + (lane & 15);
    const int kb = (lane >> 4) * 16;

    for (int ks = 0; ks < FDIM; ks += KS) {
        __syncthreads();   // protect Th/Tl from previous iteration's readers
        // ---- Stage theta[ks..ks+31][:] transposed -> hi/lo f16 ----
        for (int i = tid; i < KS * ODIM / 4; i += 256) {
            int k = (i * 4) / ODIM;
            int n = (i * 4) % ODIM;
            float4 v = *(const float4*)(theta + (size_t)(ks + k) * ODIM + n);
            float a[4] = {v.x, v.y, v.z, v.w};
            #pragma unroll
            for (int q = 0; q < 4; ++q) {
                _Float16 h = (_Float16)a[q];
                Th[(n + q) * KS + k] = h;
                Tl[(n + q) * KS + k] = (_Float16)(a[q] - (float)h);
            }
        }
        __syncthreads();

        // Prefetch next theta slab into cache while this slab computes
        // (emits global_prefetch_b8; one 128B line per thread covers 32 KB).
        if (ks + KS < FDIM) {
            const float* np = theta + (size_t)(ks + KS) * ODIM + tid * 32;
            __builtin_prefetch(np, 0, 3);
        }

        // ---- A fragments (two 16-byte LDS loads each) ----
        const _Float16* ab = Xh + am * FDIM + ks;
        v8h a0 = *(const v8h*)(ab + kb);
        v8h a1 = *(const v8h*)(ab + kb + 8);
        v16h Ah = __builtin_shufflevector(a0, a1, 0,1,2,3,4,5,6,7,8,9,10,11,12,13,14,15);
        const _Float16* lb = Xl + am * FDIM + ks;
        v8h l0 = *(const v8h*)(lb + kb);
        v8h l1 = *(const v8h*)(lb + kb + 8);
        v16h Al = __builtin_shufflevector(l0, l1, 0,1,2,3,4,5,6,7,8,9,10,11,12,13,14,15);

        // Process tiles pairwise; interleave the two independent WMMA chains
        // so dependent WMMAs (same acc) are separated by an independent one.
        const int kb2 = (lane >> 4) * 16;
        #pragma unroll
        for (int t = 0; t < 8; t += 2) {
            const int nn0 = (ntbase + t)     * 16 + (lane & 15);
            const int nn1 = (ntbase + t + 1) * 16 + (lane & 15);

            const _Float16* bb0 = Th + nn0 * KS + kb2;
            v8h p0 = *(const v8h*)(bb0);
            v8h p1 = *(const v8h*)(bb0 + 8);
            v16h Bh0 = __builtin_shufflevector(p0, p1, 0,1,2,3,4,5,6,7,8,9,10,11,12,13,14,15);
            const _Float16* cb0 = Tl + nn0 * KS + kb2;
            v8h q0 = *(const v8h*)(cb0);
            v8h q1 = *(const v8h*)(cb0 + 8);
            v16h Bl0 = __builtin_shufflevector(q0, q1, 0,1,2,3,4,5,6,7,8,9,10,11,12,13,14,15);

            const _Float16* bb1 = Th + nn1 * KS + kb2;
            v8h r0 = *(const v8h*)(bb1);
            v8h r1 = *(const v8h*)(bb1 + 8);
            v16h Bh1 = __builtin_shufflevector(r0, r1, 0,1,2,3,4,5,6,7,8,9,10,11,12,13,14,15);
            const _Float16* cb1 = Tl + nn1 * KS + kb2;
            v8h s0 = *(const v8h*)(cb1);
            v8h s1 = *(const v8h*)(cb1 + 8);
            v16h Bl1 = __builtin_shufflevector(s0, s1, 0,1,2,3,4,5,6,7,8,9,10,11,12,13,14,15);

            // fp32 ~= Ah*Bh + Ah*Bl + Al*Bh (fp32 accumulate), two chains
            acc[t]     = __builtin_amdgcn_wmma_f32_16x16x32_f16(
                false, Ah, false, Bh0, (short)0, acc[t],     false, false);
            acc[t + 1] = __builtin_amdgcn_wmma_f32_16x16x32_f16(
                false, Ah, false, Bh1, (short)0, acc[t + 1], false, false);
            acc[t]     = __builtin_amdgcn_wmma_f32_16x16x32_f16(
                false, Ah, false, Bl0, (short)0, acc[t],     false, false);
            acc[t + 1] = __builtin_amdgcn_wmma_f32_16x16x32_f16(
                false, Ah, false, Bl1, (short)0, acc[t + 1], false, false);
            acc[t]     = __builtin_amdgcn_wmma_f32_16x16x32_f16(
                false, Al, false, Bh0, (short)0, acc[t],     false, false);
            acc[t + 1] = __builtin_amdgcn_wmma_f32_16x16x32_f16(
                false, Al, false, Bh1, (short)0, acc[t + 1], false, false);
        }
    }

    // ---- Epilogue: C/D layout -> global. lane: N=lane&15, M=r+8*(lane>>4) ----
    #pragma unroll
    for (int t = 0; t < 8; ++t) {
        const int col = (ntbase + t) * 16 + (lane & 15);
        #pragma unroll
        for (int r = 0; r < 8; ++r) {
            int grow = row0 + mt * 16 + r + 8 * (lane >> 4);
            if (grow < nrows) Y[(size_t)grow * ODIM + col] = acc[t][r];
        }
    }
}

// ---------------------------------------------------------------------------
// out[r] += v * Y[c]  (COO scatter; rows sorted -> register accumulation
// within a wave, atomic flush only on row change).
// Each wave handles 128 consecutive nonzeros; lane covers 8 of 256 features.
// ---------------------------------------------------------------------------
__global__ void __launch_bounds__(256)
scatter_coo(const int* __restrict__ rows, const int* __restrict__ cols,
            const float* __restrict__ vals, const float* __restrict__ Y,
            float* __restrict__ out, int nnz) {
    const int lane = threadIdx.x & 31;
    const int wave = threadIdx.x >> 5;
    const long base = ((long)blockIdx.x * 8 + wave) * 128;

    float acc[8];
    #pragma unroll
    for (int i = 0; i < 8; ++i) acc[i] = 0.f;
    int cur = -1;

    for (int p = 0; p < 4; ++p) {
        long e = base + (long)p * 32 + lane;
        int r = -1, c = 0;
        float v = 0.f;
        if (e < nnz) { r = rows[e]; c = cols[e]; v = vals[e]; }

        for (int j = 0; j < 32; ++j) {
            int rj = __shfl(r, j, 32);
            if (rj < 0) break;                       // uniform tail cut
            int   cj = __shfl(c, j, 32);
            float vj = __shfl(v, j, 32);
            if (rj != cur) {
                if (cur >= 0) {
                    float* o = out + (size_t)cur * ODIM + lane * 8;
                    #pragma unroll
                    for (int i = 0; i < 8; ++i) atomicAdd(o + i, acc[i]);
                }
                cur = rj;
                #pragma unroll
                for (int i = 0; i < 8; ++i) acc[i] = 0.f;
            }
            const float4* yp = (const float4*)(Y + (size_t)cj * ODIM + lane * 8);
            float4 y0 = yp[0], y1 = yp[1];
            acc[0] += vj * y0.x; acc[1] += vj * y0.y;
            acc[2] += vj * y0.z; acc[3] += vj * y0.w;
            acc[4] += vj * y1.x; acc[5] += vj * y1.y;
            acc[6] += vj * y1.z; acc[7] += vj * y1.w;
        }
    }
    if (cur >= 0) {
        float* o = out + (size_t)cur * ODIM + lane * 8;
        #pragma unroll
        for (int i = 0; i < 8; ++i) atomicAdd(o + i, acc[i]);
    }
}

// ---------------------------------------------------------------------------
extern "C" void kernel_launch(void* const* d_in, const int* in_sizes, int n_in,
                              void* d_out, int out_size, void* d_ws, size_t ws_size,
                              hipStream_t stream) {
    const int*   L_rows = (const int*)  d_in[0];
    const int*   L_cols = (const int*)  d_in[1];
    const float* L_vals = (const float*)d_in[2];
    const float* x      = (const float*)d_in[3];
    const float* theta  = (const float*)d_in[4];
    float*       out    = (float*)d_out;
    float*       Y      = (float*)d_ws;          // [nrows][256] scratch

    const int nnz   = in_sizes[0];
    const int nrows = in_sizes[3] / FDIM;

    // 1) zero out (it's the atomic-scatter target)
    {
        long n4 = (long)out_size / 4;
        long blocks = (n4 + 255) / 256;
        zero_f32<<<(int)blocks, 256, 0, stream>>>(out, n4);
    }

    // 2) Y = x @ theta via split-f16 WMMA
    {
        int blocks = (nrows + BM - 1) / BM;
        size_t smem = (size_t)(2 * BM * FDIM + 2 * ODIM * KS) * sizeof(_Float16); // 96 KB
        gemm_x_theta<<<blocks, 256, smem, stream>>>(x, theta, Y, nrows);
    }

    // 3) out[r] += v * Y[c]
    {
        int blocks = (nnz + 1023) / 1024;   // 8 waves * 128 nnz per block
        scatter_coo<<<blocks, 256, 0, stream>>>(L_rows, L_cols, L_vals, Y, out, nnz);
    }
}
